// TEGNN_14508399525988
// MI455X (gfx1250) — matrix operand order repaired and checked by reading, not compiled, not run on hardware
//
#include <hip/hip_runtime.h>
#include <hip/hip_bf16.h>

// ---------------------------------------------------------------------------
// EGNN forward for MI455X (gfx1250): wave32, WMMA f32_16x16x32_f16 matrix ops.
// Grid = 256 WGs: 2 per sample (each owns 24 source rows = 1128 edges).
// Weights pre-transposed to f16 once per call; staged via coalesced b128
// copies + global_prefetch. Edge MLPs run as chunked 16x16x32 WMMA GEMMs.
// ---------------------------------------------------------------------------

#define T_SAMPLES 128
#define NN        48
#define HIDF      128
#define ROWS_L    24            // source rows per workgroup
#define EBLK      (48*47)       // 2256 edges per sample
#define EBLK_L    (ROWS_L*47)   // 1128 edges per workgroup
#define CHUNK     128           // edges per chunk (8 M-tiles)
#define NCHUNK_L  9             // ceil(1128/128)
#define WMAT      16384         // halves per 128x128 matrix
#define WLAYER    (7*WMAT)      // 7 prepped matrices per layer

typedef __attribute__((ext_vector_type(16))) _Float16 v16h;
typedef __attribute__((ext_vector_type(8)))  _Float16 v8h;
typedef __attribute__((ext_vector_type(8)))  float    v8f;

__device__ __forceinline__ float silu(float x) {
    return x / (1.0f + __expf(-x));
}

// A fragment (16x32 f16): buf row-major [M][K], base = &buf[m0*stride + k0].
__device__ __forceinline__ v16h load_a_frag(const _Float16* base, int stride, int lane) {
    const int row  = lane & 15;
    const int koff = (lane >> 4) ? 8 : 0;
    const _Float16* p = base + row * stride + koff;
    v8h a0 = *(const v8h*)(p);
    v8h a1 = *(const v8h*)(p + 16);
    v16h A;
#pragma unroll
    for (int i = 0; i < 8; ++i) { A[i] = a0[i]; A[i + 8] = a1[i]; }
    return A;
}

// B fragment (32x16 f16): weights stored transposed Wt[n][k], base = &Wt[n0*stride + k0].
__device__ __forceinline__ v16h load_b_frag(const _Float16* base, int stride, int lane) {
    const int n    = lane & 15;
    const int koff = (lane >> 4) ? 8 : 0;
    const _Float16* p = base + n * stride + koff;
    v8h b0 = *(const v8h*)(p);
    v8h b1 = *(const v8h*)(p + 16);
    v16h B;
#pragma unroll
    for (int i = 0; i < 8; ++i) { B[i] = b0[i]; B[i + 8] = b1[i]; }
    return B;
}

__device__ __forceinline__ v8f wmma_f16(v16h a, v16h b, v8f c) {
    return __builtin_amdgcn_wmma_f32_16x16x32_f16(
        false, a, false, b, (short)0, c, false, false);
}

// Coalesced 128x128 f16 matrix copy global -> LDS (b128 loads / ds_store_b128).
__device__ __forceinline__ void copy_w(_Float16* dst, const _Float16* src, int tid) {
    const v8h* s = (const v8h*)src;
    v8h* d = (v8h*)dst;
#pragma unroll
    for (int i = 0; i < 8; ++i) d[tid + 256 * i] = s[tid + 256 * i];
}

// ---------------------------------------------------------------------------
// Kernel 0: weight prep (once per call). Transpose + f16-convert per layer:
// [0]=W1a^T [1]=W1b^T [2]=W2^T [3]=C1^T [4]=N1a^T [5]=N1b^T [6]=N2^T
// ---------------------------------------------------------------------------
__global__ __launch_bounds__(256) void egnn_wprep_kernel(
    const float* __restrict__ e1w, const float* __restrict__ e2w,
    const float* __restrict__ c1w, const float* __restrict__ n1w,
    const float* __restrict__ n2w, _Float16* __restrict__ dst) {
    const int idx = blockIdx.x * 256 + threadIdx.x;
    if (idx >= WLAYER) return;
    const int m = idx >> 14;
    const int i = idx & (WMAT - 1);
    const int n = i >> 7, k = i & 127;
    const float* src;
    int koff = 0;
    switch (m) {
        case 0: src = e1w; break;
        case 1: src = e1w; koff = 128; break;
        case 2: src = e2w; break;
        case 3: src = c1w; break;
        case 4: src = n1w; break;
        case 5: src = n1w; koff = 128; break;
        default: src = n2w; break;
    }
    dst[idx] = (_Float16)src[(k + koff) * 128 + n];
}

// ---------------------------------------------------------------------------
// Kernel 1: timestep embedding MLP.  t[T] -> temb[T][128].
// ---------------------------------------------------------------------------
__global__ __launch_bounds__(128) void egnn_temb_kernel(
    const float* __restrict__ t, const float* __restrict__ W1, const float* __restrict__ b1,
    const float* __restrict__ W2, const float* __restrict__ b2, float* __restrict__ temb) {
    __shared__ float emb[256];
    __shared__ float hid[128];
    const int ti = blockIdx.x;
    const int j  = threadIdx.x;
    const float tv = t[ti];
    const float fr = __expf(-logf(10000.f) * (float)j * (1.0f / 128.0f));
    const float a  = tv * fr;
    emb[j]       = cosf(a);
    emb[128 + j] = sinf(a);
    __syncthreads();
    float acc = b1[j];
    for (int k = 0; k < 256; ++k) acc += emb[k] * W1[k * 128 + j];
    hid[j] = silu(acc);
    __syncthreads();
    float acc2 = b2[j];
    for (int k = 0; k < 128; ++k) acc2 += hid[k] * W2[k * 128 + j];
    temb[ti * 128 + j] = acc2;
}

// ---------------------------------------------------------------------------
// Kernel 2: node embedding broadcast.  h[N][8] @ W[8][128] + b -> hstate[TN][128].
// ---------------------------------------------------------------------------
__global__ __launch_bounds__(256) void egnn_hinit_kernel(
    const float* __restrict__ h, const float* __restrict__ W, const float* __restrict__ b,
    float* __restrict__ hstate) {
    const int idx = blockIdx.x * 256 + threadIdx.x;
    if (idx >= T_SAMPLES * NN * HIDF) return;
    const int j = idx & 127;
    const int node = (idx >> 7) % NN;
    float acc = b[j];
#pragma unroll
    for (int k = 0; k < 8; ++k) acc += h[node * 8 + k] * W[k * 128 + j];
    hstate[idx] = acc;
}

// ---------------------------------------------------------------------------
// Kernel 3: one EGNN layer. grid.x = 2*T; WG owns rows r0..r0+23 of sample t.
// ---------------------------------------------------------------------------
#define SMEM_BYTES 188928

__global__ __launch_bounds__(256, 1) void egnn_layer_kernel(
    float* __restrict__ hstate, float* __restrict__ xstate,
    const float* __restrict__ temb, const float* __restrict__ edge_attr,
    const _Float16* __restrict__ wt,                       // prepped 7x[128][128] f16
    const float* __restrict__ e1w, const float* __restrict__ e1b,
    const float* __restrict__ e2b, const float* __restrict__ n1b,
    const float* __restrict__ n2b, const float* __restrict__ c1b,
    const float* __restrict__ c2w) {
    extern __shared__ char smem[];
    _Float16* sH   = (_Float16*)(smem);            // [48][128] h(+temb) f16   12288 B
    _Float16* sP   = (_Float16*)(smem + 12288);    // [48][128] P (f16)        12288 B
    _Float16* sQ   = (_Float16*)(smem + 24576);    // [48][128] Q (f16)        12288 B
    _Float16* sW   = (_Float16*)(smem + 36864);    // 2x [128][128] Wt f16     65536 B
    _Float16* sA   = (_Float16*)(smem + 102400);   // [128][128] e1 chunk      32768 B
    _Float16* sB   = (_Float16*)(smem + 135168);   // [128][128] ef chunk      32768 B
    float*    sAgg = (float*)(smem + 167936);      // [24][128] agg f32        12288 B
    float*    sXl  = (float*)(smem + 180224);      // [48][4]                    768 B
    float*    sXa  = (float*)(smem + 180992);      // [24][4] (+pad)             768 B
    int*      sRow = (int*)(smem + 181760);        // [128] local row            512 B
    float*    sCv  = (float*)(smem + 182272);      // [128] coord scalar         512 B
    float*    sDif = (float*)(smem + 182784);      // [128][4] norm diff        2048 B
    float*    sBia = (float*)(smem + 184832);      // 8 x [128] bias vecs       4096 B

    float* bb1 = sBia;       float* bb2 = sBia + 128;
    float* bc1 = sBia + 256; float* bcv = sBia + 384;
    float* bwr = sBia + 512; float* bwa = sBia + 640;
    float* bn1 = sBia + 768; float* bn2 = sBia + 896;

    const int tid  = threadIdx.x;
    const int lane = tid & 31;
    const int wave = tid >> 5;
    const int t    = blockIdx.x >> 1;
    const int r0   = (blockIdx.x & 1) * ROWS_L;

    // ---- Phase 1: stage h(+temb) f16 (all 48 nodes), x, biases ------------
    for (int i = tid; i < NN * HIDF; i += 256) {
        const int n = i >> 7, j = i & 127;
        const float v = hstate[(t * NN + n) * HIDF + j] + temb[t * HIDF + j];
        sH[i] = (_Float16)v;
    }
    for (int i = tid; i < ROWS_L * HIDF; i += 256) sAgg[i] = 0.0f;
    if (tid < 128) {
        bb1[tid] = e1b[tid];
        bb2[tid] = e2b[tid];
        bc1[tid] = c1b[tid];
        bcv[tid] = c2w[tid];
        bwr[tid] = e1w[256 * 128 + tid];   // radial row of edge1.w
        bwa[tid] = e1w[257 * 128 + tid];   // edge_attr row of edge1.w
        bn1[tid] = n1b[tid];
        bn2[tid] = n2b[tid];
    }
    for (int i = tid; i < NN * 4; i += 256) {
        const int r = i >> 2, d = i & 3;
        sXl[i] = (d < 3) ? xstate[(t * NN + r) * 3 + d] : 0.0f;
    }
    for (int i = tid; i < ROWS_L * 4; i += 256) sXa[i] = 0.0f;
    // warm L2 for the edge-phase weights while we stage activations
    __builtin_prefetch(wt + 0 * WMAT + tid * 64, 0, 1);
    __builtin_prefetch(wt + 1 * WMAT + tid * 64, 0, 1);
    __syncthreads();

    // ---- Phase 2: stage W1a^T, W1b^T (coalesced b128 copies) --------------
    copy_w(sW,         wt + 0 * WMAT, tid);
    copy_w(sW + WMAT,  wt + 1 * WMAT, tid);
    __builtin_prefetch(wt + 2 * WMAT + tid * 64, 0, 1);
    __builtin_prefetch(wt + 3 * WMAT + tid * 64, 0, 1);
    __syncthreads();

    // ---- Phase 3: P = h@W1a + b1 (own 24 rows), Q = h@W1b (all 48 rows) ---
    for (int job = wave; job < 40; job += 8) {
        v8f acc = {};
        if (job < 16) {                       // P: 2 M-tiles x 8 N-tiles
            const int mt = job & 1, nt = job >> 1;
#pragma unroll
            for (int kt = 0; kt < 4; ++kt) {
                v16h Af = load_a_frag(sH + (r0 + mt * 16) * HIDF + kt * 32, HIDF, lane);
                v16h Bf = load_b_frag(sW + (nt * 16) * HIDF + kt * 32, HIDF, lane);
                acc = wmma_f16(Af, Bf, acc);
            }
            const int col   = nt * 16 + (lane & 15);
            const int rbase = mt * 16 + ((lane >> 4) << 3);
#pragma unroll
            for (int v = 0; v < 8; ++v)
                if (rbase + v < ROWS_L)
                    sP[(r0 + rbase + v) * HIDF + col] = (_Float16)(acc[v] + bb1[col]);
        } else {                              // Q: 3 M-tiles x 8 N-tiles
            const int jq = job - 16;
            const int mt = jq % 3, nt = jq / 3;
#pragma unroll
            for (int kt = 0; kt < 4; ++kt) {
                v16h Af = load_a_frag(sH + (mt * 16) * HIDF + kt * 32, HIDF, lane);
                v16h Bf = load_b_frag(sW + WMAT + (nt * 16) * HIDF + kt * 32, HIDF, lane);
                acc = wmma_f16(Af, Bf, acc);
            }
            const int col   = nt * 16 + (lane & 15);
            const int rbase = mt * 16 + ((lane >> 4) << 3);
#pragma unroll
            for (int v = 0; v < 8; ++v)
                sQ[(rbase + v) * HIDF + col] = (_Float16)acc[v];
        }
    }
    __syncthreads();

    // ---- Phase 4: stage W2^T (edge2) and C1^T (coord1) --------------------
    copy_w(sW,        wt + 2 * WMAT, tid);
    copy_w(sW + WMAT, wt + 3 * WMAT, tid);
    __builtin_prefetch(wt + 4 * WMAT + tid * 64, 0, 1);
    __builtin_prefetch(wt + 5 * WMAT + tid * 64, 0, 1);
    __builtin_prefetch(wt + 6 * WMAT + tid * 64, 0, 1);
    __syncthreads();

    // ---- Phase 5: edge chunks (1128 edges, 9 chunks of 128) ---------------
    for (int ch = 0; ch < NCHUNK_L; ++ch) {
        const int e0 = ch * CHUNK;
        // (a) stage e1 = silu(P[r] + Q[c] + radial*wr + ea*wa)
        {
            const int s = tid >> 1, half = tid & 1;
            const int e = e0 + s;
            const int j0 = half * 64;
            if (e < EBLK_L) {
                const int rl = e / 47;
                const int q  = e - rl * 47;
                const int r  = r0 + rl;
                const int c  = q + (q >= r);
                const float dx = sXl[r * 4 + 0] - sXl[c * 4 + 0];
                const float dy = sXl[r * 4 + 1] - sXl[c * 4 + 1];
                const float dz = sXl[r * 4 + 2] - sXl[c * 4 + 2];
                const float rad = dx * dx + dy * dy + dz * dz;
                const float inv = rsqrtf(rad + 1e-8f);
                const float ea = edge_attr[t * EBLK + r * 47 + q];
                if (half == 0) {
                    sRow[s] = rl;
                    sCv[s] = 0.0f;
                    sDif[s * 4 + 0] = dx * inv;
                    sDif[s * 4 + 1] = dy * inv;
                    sDif[s * 4 + 2] = dz * inv;
                    sDif[s * 4 + 3] = rad;
                }
                for (int j = j0; j < j0 + 64; ++j) {
                    const float v = (float)sP[r * HIDF + j] + (float)sQ[c * HIDF + j]
                                  + rad * bwr[j] + ea * bwa[j];
                    sA[s * HIDF + j] = (_Float16)silu(v);
                }
            } else {
                if (half == 0) {
                    sRow[s] = -1;
                    sCv[s] = 0.0f;
                    sDif[s * 4 + 0] = 0.f; sDif[s * 4 + 1] = 0.f;
                    sDif[s * 4 + 2] = 0.f; sDif[s * 4 + 3] = 0.f;
                }
                for (int j = j0; j < j0 + 64; ++j) sA[s * HIDF + j] = (_Float16)0.0f;
            }
        }
        __syncthreads();

        // (b) ef = silu(e1 @ W2 + b2); agg[r] += ef
        for (int job = wave; job < 64; job += 8) {
            const int mt = job & 7, nt = job >> 3;
            v8f acc = {};
#pragma unroll
            for (int kt = 0; kt < 4; ++kt) {
                v16h Af = load_a_frag(sA + (mt * 16) * HIDF + kt * 32, HIDF, lane);
                v16h Bf = load_b_frag(sW + (nt * 16) * HIDF + kt * 32, HIDF, lane);
                acc = wmma_f16(Af, Bf, acc);
            }
            const int col   = nt * 16 + (lane & 15);
            const int rbase = mt * 16 + ((lane >> 4) << 3);
#pragma unroll
            for (int v = 0; v < 8; ++v) {
                const float val = silu(acc[v] + bb2[col]);
                sB[(rbase + v) * HIDF + col] = (_Float16)val;
                const int rl = sRow[rbase + v];
                if (rl >= 0) atomicAdd(&sAgg[rl * HIDF + col], val);
            }
        }
        __syncthreads();

        // (c) coordval = silu(ef @ C1 + c1b) . c2w  (coord2 fused as dot)
        for (int job = wave; job < 64; job += 8) {
            const int mt = job & 7, nt = job >> 3;
            v8f acc = {};
#pragma unroll
            for (int kt = 0; kt < 4; ++kt) {
                v16h Af = load_a_frag(sB + (mt * 16) * HIDF + kt * 32, HIDF, lane);
                v16h Bf = load_b_frag(sW + WMAT + (nt * 16) * HIDF + kt * 32, HIDF, lane);
                acc = wmma_f16(Af, Bf, acc);
            }
            const int col   = nt * 16 + (lane & 15);
            const int rbase = mt * 16 + ((lane >> 4) << 3);
#pragma unroll
            for (int v = 0; v < 8; ++v) {
                const int s = rbase + v;
                if (sRow[s] >= 0) {
                    const float val = silu(acc[v] + bc1[col]) * bcv[col];
                    atomicAdd(&sCv[s], val);
                }
            }
        }
        __syncthreads();

        // (d) xacc[r] += diffn * coordval
        if (tid < CHUNK) {
            const int s = tid;
            const int rl = sRow[s];
            if (rl >= 0) {
                const float cv = sCv[s];
                atomicAdd(&sXa[rl * 4 + 0], sDif[s * 4 + 0] * cv);
                atomicAdd(&sXa[rl * 4 + 1], sDif[s * 4 + 1] * cv);
                atomicAdd(&sXa[rl * 4 + 2], sDif[s * 4 + 2] * cv);
            }
        }
        __syncthreads();
    }

    // ---- Phase 6: x writeback; agg->f16 (sP rows 0..23); stage N1a^T,N1b^T
    if (tid < ROWS_L * 3) {
        const int rl = tid / 3, d = tid % 3;
        xstate[(t * NN + r0 + rl) * 3 + d] = sXl[(r0 + rl) * 4 + d] + sXa[rl * 4 + d];
    }
    for (int i = tid; i < ROWS_L * HIDF; i += 256) sP[i] = (_Float16)sAgg[i];
    copy_w(sW,        wt + 4 * WMAT, tid);
    copy_w(sW + WMAT, wt + 5 * WMAT, tid);
    __syncthreads();

    // ---- Phase 7a: n1 = silu(h@N1a + agg@N1b + b) -> sA (local rows) ------
    for (int job = wave; job < 16; job += 8) {
        const int mt = job & 1, nt = job >> 1;
        v8f acc = {};
#pragma unroll
        for (int kt = 0; kt < 4; ++kt) {
            v16h Af = load_a_frag(sH + (r0 + mt * 16) * HIDF + kt * 32, HIDF, lane);
            v16h Bf = load_b_frag(sW + (nt * 16) * HIDF + kt * 32, HIDF, lane);
            acc = wmma_f16(Af, Bf, acc);
        }
#pragma unroll
        for (int kt = 0; kt < 4; ++kt) {
            v16h Af = load_a_frag(sP + (mt * 16) * HIDF + kt * 32, HIDF, lane);
            v16h Bf = load_b_frag(sW + WMAT + (nt * 16) * HIDF + kt * 32, HIDF, lane);
            acc = wmma_f16(Af, Bf, acc);
        }
        const int col   = nt * 16 + (lane & 15);
        const int rbase = mt * 16 + ((lane >> 4) << 3);
#pragma unroll
        for (int v = 0; v < 8; ++v)
            sA[(rbase + v) * HIDF + col] = (_Float16)silu(acc[v] + bn1[col]);
    }
    __syncthreads();

    // ---- Phase 7b: stage N2^T; h_new = h + temb + n1@N2 + b ---------------
    copy_w(sW, wt + 6 * WMAT, tid);
    __syncthreads();
    for (int job = wave; job < 16; job += 8) {
        const int mt = job & 1, nt = job >> 1;
        v8f acc = {};
#pragma unroll
        for (int kt = 0; kt < 4; ++kt) {
            v16h Af = load_a_frag(sA + (mt * 16) * HIDF + kt * 32, HIDF, lane);
            v16h Bf = load_b_frag(sW + (nt * 16) * HIDF + kt * 32, HIDF, lane);
            acc = wmma_f16(Af, Bf, acc);
        }
        const int col   = nt * 16 + (lane & 15);
        const int rbase = mt * 16 + ((lane >> 4) << 3);
#pragma unroll
        for (int v = 0; v < 8; ++v) {
            const int rl = rbase + v;
            if (rl < ROWS_L) {
                const int g = (t * NN + r0 + rl) * HIDF + col;
                const float hl = hstate[g] + temb[t * HIDF + col];
                hstate[g] = hl + acc[v] + bn2[col];
            }
        }
    }
}

// ---------------------------------------------------------------------------
// Kernel 4: output head.  hstate[TN][128] @ Wout[128][8] + b -> hout[TN][8].
// ---------------------------------------------------------------------------
__global__ __launch_bounds__(256) void egnn_outhead_kernel(
    const float* __restrict__ hstate, const float* __restrict__ W,
    const float* __restrict__ b, float* __restrict__ hout) {
    const int idx = blockIdx.x * 256 + threadIdx.x;
    if (idx >= T_SAMPLES * NN * 8) return;
    const int o = idx & 7, row = idx >> 3;
    float acc = b[o];
    for (int k = 0; k < 128; ++k) acc += hstate[row * 128 + k] * W[k * 8 + o];
    hout[idx] = acc;
}

// ---------------------------------------------------------------------------
extern "C" void kernel_launch(void* const* d_in, const int* in_sizes, int n_in,
                              void* d_out, int out_size, void* d_ws, size_t ws_size,
                              hipStream_t stream) {
    (void)in_sizes; (void)n_in; (void)out_size; (void)ws_size;
    const float* h_in  = (const float*)d_in[0];
    const float* x_in  = (const float*)d_in[1];
    const float* t_in  = (const float*)d_in[2];
    const float* ea_in = (const float*)d_in[3];
    // d_in[4] = edges (int32) -- deterministic structure, recomputed on device
    const float* embW = (const float*)d_in[5];
    const float* embB = (const float*)d_in[6];
    const float* outW = (const float*)d_in[7];
    const float* outB = (const float*)d_in[8];
    const float* t1W  = (const float*)d_in[9];
    const float* t1B  = (const float*)d_in[10];
    const float* t2W  = (const float*)d_in[11];
    const float* t2B  = (const float*)d_in[12];

    float* hout = (float*)d_out;                               // [TN][8]
    float* xout = (float*)d_out + T_SAMPLES * NN * 8;          // [TN][3] (x state)
    float* temb = (float*)d_ws;                                // [T][128]
    float* hst  = (float*)d_ws + T_SAMPLES * 128;              // [TN][128]
    _Float16* wt16 = (_Float16*)((char*)d_ws + 65536 + 3145728);  // 4 x 7 x 128x128 f16

    // weight prep (transposed f16) for all 4 layers
    for (int l = 0; l < 4; ++l) {
        const int base = 13 + 11 * l;
        egnn_wprep_kernel<<<WLAYER / 256, 256, 0, stream>>>(
            (const float*)d_in[base + 0],   // edge1.w
            (const float*)d_in[base + 2],   // edge2.w
            (const float*)d_in[base + 8],   // coord1.w
            (const float*)d_in[base + 4],   // node1.w
            (const float*)d_in[base + 6],   // node2.w
            wt16 + l * WLAYER);
    }
    // t embedding MLP
    egnn_temb_kernel<<<T_SAMPLES, 128, 0, stream>>>(t_in, t1W, t1B, t2W, t2B, temb);
    // h state init (broadcast node embedding across T)
    egnn_hinit_kernel<<<(T_SAMPLES * NN * HIDF) / 256, 256, 0, stream>>>(h_in, embW, embB, hst);
    // x state init
    hipMemcpyAsync(xout, x_in, (size_t)T_SAMPLES * NN * 3 * sizeof(float),
                   hipMemcpyDeviceToDevice, stream);

    for (int l = 0; l < 4; ++l) {
        const int base = 13 + 11 * l;
        egnn_layer_kernel<<<2 * T_SAMPLES, 256, SMEM_BYTES, stream>>>(
            hst, xout, temb, ea_in,
            wt16 + l * WLAYER,
            (const float*)d_in[base + 0],   // edge1.w (radial/ea rows)
            (const float*)d_in[base + 1],   // edge1.b
            (const float*)d_in[base + 3],   // edge2.b
            (const float*)d_in[base + 5],   // node1.b
            (const float*)d_in[base + 7],   // node2.b
            (const float*)d_in[base + 9],   // coord1.b
            (const float*)d_in[base + 10]); // coord2.w
    }

    egnn_outhead_kernel<<<(T_SAMPLES * NN * 8) / 256, 256, 0, stream>>>(hst, outW, outB, hout);
}
